// ForceModule_21904333209925
// MI455X (gfx1250) — matrix operand order repaired and checked by reading, not compiled, not run on hardware
//
#include <hip/hip_runtime.h>
#include <hip/hip_bf16.h>

typedef __attribute__((ext_vector_type(16))) _Float16 v16h;
typedef __attribute__((ext_vector_type(8)))  float    v8f;

#define NA 16384   // total atoms
#define NM 8192    // filtered rows
#define ND 64      // feature width

// ---------------------------------------------------------------------------
// Kernel 0: hT[n][k] = (f16) embed[node_types[k]][n]   (transposed feature table)
// ---------------------------------------------------------------------------
__global__ __launch_bounds__(256) void build_hT(const float* __restrict__ embed,
                                                const int* __restrict__ node_types,
                                                _Float16* __restrict__ hT) {
    int tid = blockIdx.x * 256 + threadIdx.x;     // 0 .. ND*NA-1
    int k = tid & (NA - 1);
    int n = tid >> 14;                            // tid / NA
    hT[(size_t)n * NA + k] = (_Float16)embed[node_types[k] * ND + n];
}

// distance weight on squared distance: mask on d2 (equivalent to reference:
// 0.01 < d <= 0.5  <=>  1e-4 < d2 <= 0.25; the 1e-12 clamp lies inside the
// masked-out region). Keeps the compares off the sqrt/exp critical path.
__device__ __forceinline__ float wfun(float px, float py, float pz, float4 q) {
    float dx = px - q.x, dy = py - q.y, dz = pz - q.z;
    float d2 = fmaf(dx, dx, fmaf(dy, dy, dz * dz));
    float w  = __expf(-10.0f * sqrtf(d2));
    return (d2 > 1e-4f && d2 <= 0.25f) ? w : 0.0f;
}

// Build the 16x32 f16 A-fragment (contiguous k = hi*16 + s order, matching B) from a staged 32-atom tile.
// VGPR pair p : K = hi*16 + 2p, +1
__device__ __forceinline__ v16h make_a(const float4* __restrict__ sp, int kb,
                                       float px, float py, float pz) {
    v16h a;
#pragma unroll
    for (int p = 0; p < 8; ++p) {
        int kk = kb + 2 * p;
        float4 q0 = sp[kk];
        float4 q1 = sp[kk + 1];
        a[2 * p + 0] = (_Float16)wfun(px, py, pz, q0);
        a[2 * p + 1] = (_Float16)wfun(px, py, pz, q1);
    }
    return a;
}

// ---------------------------------------------------------------------------
// Kernel 1: fused w-generation + agg = w @ h via v_wmma_f32_16x16x32_f16.
// One wave per 16-row output tile; 4 waves (64 rows) per workgroup.
// 2x-unrolled software pipeline with ping-pong register sets + LDS buffers:
// no register rotation copies, loads get a half-iteration of latency slack.
// ---------------------------------------------------------------------------
__global__ __launch_bounds__(128) void agg_wmma(const float* __restrict__ pos,
                                                const int* __restrict__ filter_idx,
                                                const _Float16* __restrict__ hT,
                                                float* __restrict__ agg) {
    __shared__ float4 spos[2][32];

    const int lane = threadIdx.x & 31;
    const int wave = threadIdx.x >> 5;
    const int ln   = lane & 15;      // column-within-tile for B/C/D
    const int hi   = lane >> 4;      // half-wave selector
    const int kb   = hi * 16;
    const int m0   = (blockIdx.x * 4 + wave) * 16;

    // this lane's output row atom (lanes l and l+16 share row m0+ln)
    const int   row = filter_idx[m0 + ln];
    const float px = pos[row * 3 + 0];
    const float py = pos[row * 3 + 1];
    const float pz = pos[row * 3 + 2];

    v8f acc0 = {}, acc1 = {}, acc2 = {}, acc3 = {};

    // B base pointer for this lane: hT row (16*b + ln), K offset hi*16
    const _Float16* bp = hT + (size_t)ln * NA + hi * 16;

    // ---- prologue: stage chunk 0 positions + B fragment set c ----
    if (threadIdx.x < 32) {
        int j = threadIdx.x;
        spos[0][threadIdx.x] = make_float4(pos[j * 3 + 0], pos[j * 3 + 1],
                                           pos[j * 3 + 2], 0.0f);
    }
    v16h c0 = *(const v16h*)(bp + (size_t)0  * NA);
    v16h c1 = *(const v16h*)(bp + (size_t)16 * NA);
    v16h c2 = *(const v16h*)(bp + (size_t)32 * NA);
    v16h c3 = *(const v16h*)(bp + (size_t)48 * NA);
    __syncthreads();

    for (int k0 = 0; k0 < NA; k0 += 64) {
        v16h d0, d1, d2, d3;
        // ===== half 0: compute chunk k0 (set c, buf 0); prefetch k0+32 =====
        {
            const int k1 = k0 + 32;
            if (threadIdx.x < 32) {
                int j = k1 + threadIdx.x;
                spos[1][threadIdx.x] = make_float4(pos[j * 3 + 0], pos[j * 3 + 1],
                                                   pos[j * 3 + 2], 0.0f);
            }
            const _Float16* bpn = bp + k1;
            d0 = *(const v16h*)(bpn + (size_t)0  * NA);
            d1 = *(const v16h*)(bpn + (size_t)16 * NA);
            d2 = *(const v16h*)(bpn + (size_t)32 * NA);
            d3 = *(const v16h*)(bpn + (size_t)48 * NA);

            v16h a = make_a(spos[0], kb, px, py, pz);

            acc0 = __builtin_amdgcn_wmma_f32_16x16x32_f16(false, a, false, c0,
                                                          (short)0, acc0, false, false);
            acc1 = __builtin_amdgcn_wmma_f32_16x16x32_f16(false, a, false, c1,
                                                          (short)0, acc1, false, false);
            acc2 = __builtin_amdgcn_wmma_f32_16x16x32_f16(false, a, false, c2,
                                                          (short)0, acc2, false, false);
            acc3 = __builtin_amdgcn_wmma_f32_16x16x32_f16(false, a, false, c3,
                                                          (short)0, acc3, false, false);
            __syncthreads();
        }
        // ===== half 1: compute chunk k0+32 (set d, buf 1); prefetch k0+64 =====
        {
            const int k1 = (k0 + 64 < NA) ? (k0 + 64) : 0;   // wrap harmlessly
            if (threadIdx.x < 32) {
                int j = k1 + threadIdx.x;
                spos[0][threadIdx.x] = make_float4(pos[j * 3 + 0], pos[j * 3 + 1],
                                                   pos[j * 3 + 2], 0.0f);
            }
            const _Float16* bpn = bp + k1;
            c0 = *(const v16h*)(bpn + (size_t)0  * NA);
            c1 = *(const v16h*)(bpn + (size_t)16 * NA);
            c2 = *(const v16h*)(bpn + (size_t)32 * NA);
            c3 = *(const v16h*)(bpn + (size_t)48 * NA);

            v16h a = make_a(spos[1], kb, px, py, pz);

            acc0 = __builtin_amdgcn_wmma_f32_16x16x32_f16(false, a, false, d0,
                                                          (short)0, acc0, false, false);
            acc1 = __builtin_amdgcn_wmma_f32_16x16x32_f16(false, a, false, d1,
                                                          (short)0, acc1, false, false);
            acc2 = __builtin_amdgcn_wmma_f32_16x16x32_f16(false, a, false, d2,
                                                          (short)0, acc2, false, false);
            acc3 = __builtin_amdgcn_wmma_f32_16x16x32_f16(false, a, false, d3,
                                                          (short)0, acc3, false, false);
            __syncthreads();
        }
    }

    // C/D layout: VGPR g -> M = m0 + g + 8*hi, N = ln (+16 per accumulator block)
#pragma unroll
    for (int g = 0; g < 8; ++g) {
        float* dst = agg + (size_t)(m0 + g + 8 * hi) * ND + ln;
        dst[0]  = acc0[g];
        dst[16] = acc1[g];
        dst[32] = acc2[g];
        dst[48] = acc3[g];
    }
}

// ---------------------------------------------------------------------------
// Kernel 2: per-row MLP + squared error. One wave per row; lane handles
// features n and n+32. pred2[m] = (z@W3 - target)^2
// ---------------------------------------------------------------------------
__global__ __launch_bounds__(256) void mlp_rows(const float* __restrict__ embed,
                                                const int* __restrict__ node_types,
                                                const int* __restrict__ filter_idx,
                                                const float* __restrict__ W1,
                                                const float* __restrict__ W2,
                                                const float* __restrict__ b1,
                                                const float* __restrict__ W3,
                                                const float* __restrict__ target,
                                                const float* __restrict__ agg,
                                                float* __restrict__ pred2) {
    const int lane = threadIdx.x & 31;
    const int wave = threadIdx.x >> 5;
    const int m    = blockIdx.x * 8 + wave;

    const int atom = filter_idx[m];
    const float* hrow = embed + (size_t)node_types[atom] * ND;
    const float* arow = agg + (size_t)m * ND;

    const int n0 = lane, n1 = lane + 32;
    float z0 = b1[n0], z1 = b1[n1];
#pragma unroll 4
    for (int k = 0; k < ND; ++k) {
        float hk = hrow[k];
        float ak = arow[k];
        const float* w1r = W1 + (size_t)k * ND;
        const float* w2r = W2 + (size_t)k * ND;
        z0 = fmaf(hk, w1r[n0], z0);
        z0 = fmaf(ak, w2r[n0], z0);
        z1 = fmaf(hk, w1r[n1], z1);
        z1 = fmaf(ak, w2r[n1], z1);
    }
    z0 = fmaxf(z0, 0.0f);
    z1 = fmaxf(z1, 0.0f);
    float p = fmaf(z0, W3[n0], z1 * W3[n1]);

    // wave32 reduction
    for (int off = 16; off > 0; off >>= 1)
        p += __shfl_down(p, off, 32);

    if (lane == 0) {
        float diff = p - target[m];
        pred2[m] = diff * diff;
    }
}

// ---------------------------------------------------------------------------
// Kernel 3: deterministic single-block reduction of pred2 -> out[0]
// ---------------------------------------------------------------------------
__global__ __launch_bounds__(256) void reduce_sum(const float* __restrict__ pred2,
                                                  float* __restrict__ out) {
    __shared__ float s[256];
    float acc = 0.0f;
    for (int i = threadIdx.x; i < NM; i += 256)
        acc += pred2[i];
    s[threadIdx.x] = acc;
    __syncthreads();
    for (int off = 128; off > 0; off >>= 1) {
        if (threadIdx.x < off) s[threadIdx.x] += s[threadIdx.x + off];
        __syncthreads();
    }
    if (threadIdx.x == 0) out[0] = s[0];
}

// ---------------------------------------------------------------------------
extern "C" void kernel_launch(void* const* d_in, const int* in_sizes, int n_in,
                              void* d_out, int out_size, void* d_ws, size_t ws_size,
                              hipStream_t stream) {
    const float* positions  = (const float*)d_in[0];
    const float* embed      = (const float*)d_in[1];
    const float* W1         = (const float*)d_in[2];
    const float* W2         = (const float*)d_in[3];
    const float* b1         = (const float*)d_in[4];
    const float* W3         = (const float*)d_in[5];
    const float* target_cs  = (const float*)d_in[6];
    const int*   node_types = (const int*)d_in[7];
    const int*   filter_idx = (const int*)d_in[8];

    char* ws = (char*)d_ws;
    _Float16* hT    = (_Float16*)ws;                                     // 2 MB
    float*    agg   = (float*)(ws + (size_t)ND * NA * sizeof(_Float16)); // 2 MB
    float*    pred2 = (float*)(ws + (size_t)ND * NA * sizeof(_Float16)
                                  + (size_t)NM * ND * sizeof(float));    // 32 KB

    build_hT<<<(ND * NA) / 256, 256, 0, stream>>>(embed, node_types, hT);
    agg_wmma<<<NM / 64, 128, 0, stream>>>(positions, filter_idx, hT, agg);
    mlp_rows<<<NM / 8, 256, 0, stream>>>(embed, node_types, filter_idx,
                                         W1, W2, b1, W3, target_cs, agg, pred2);
    reduce_sum<<<1, 256, 0, stream>>>(pred2, (float*)d_out);
}
